// SAAR_50156628082716
// MI455X (gfx1250) — matrix operand order repaired and checked by reading, not compiled, not run on hardware
//
#include <hip/hip_runtime.h>
#include <hip/hip_bf16.h>

// ---------------------------------------------------------------------------
// Types for CDNA5 WMMA (gfx1250, wave32)
// ---------------------------------------------------------------------------
typedef __attribute__((ext_vector_type(16))) __bf16 v16bf;
typedef __attribute__((ext_vector_type(8)))  __bf16 v8bf;
typedef __attribute__((ext_vector_type(8)))  float  v8f;

union ABfrag { v16bf v; v8bf h[2]; };

__device__ __forceinline__ unsigned short f2bf(float f) {
    unsigned u = __float_as_uint(f);
    unsigned r = u + 0x7fffu + ((u >> 16) & 1u);   // round-to-nearest-even
    return (unsigned short)(r >> 16);
}

__device__ __forceinline__ float sigm(float x) {
    return 1.0f / (1.0f + __expf(-x));
}

// ---------------------------------------------------------------------------
// Kernel 0: init deg=1 (self-loop), zero agg buffers
// ---------------------------------------------------------------------------
__global__ void k_init(float* deg, float* agg1, float* agg2, int N) {
    int i = blockIdx.x * blockDim.x + threadIdx.x;
    if (i < N) deg[i] = 1.0f;
    if (i < N * 10) agg1[i] = 0.0f;
    if (i < N * 128) agg2[i] = 0.0f;
}

// Kernel 1: W_ih f32 -> bf16 (L2-resident B matrix for WMMA)
__global__ void k_w2bf(const float* __restrict__ W, unsigned short* __restrict__ Wb, int n) {
    int i = blockIdx.x * blockDim.x + threadIdx.x;
    if (i < n) Wb[i] = f2bf(W[i]);
}

// Kernel 2: degree accumulation over dst
__global__ void k_deg(const int* __restrict__ eidx, int E, float* deg) {
    int e = blockIdx.x * blockDim.x + threadIdx.x;
    if (e < E) atomicAdd(&deg[eidx[E + e]], 1.0f);
}

// Kernel 3: deg -> rsqrt(deg) in place (deg >= 1 always due to self loops)
__global__ void k_dinv(float* deg, int N) {
    int i = blockIdx.x * blockDim.x + threadIdx.x;
    if (i < N) {
        float d = deg[i];
        deg[i] = (d > 0.0f) ? rsqrtf(d) : 0.0f;
    }
}

// ---------------------------------------------------------------------------
// Kernel 4: h1 = x @ W1   (15000x768 @ 768x10) — memory-bound, block per node
// ---------------------------------------------------------------------------
__global__ void k_xw1(const float* __restrict__ x, const float* __restrict__ W1,
                      float* __restrict__ h1) {
    int n = blockIdx.x;
    int tid = threadIdx.x;              // 256 threads
    float acc[10];
#pragma unroll
    for (int j = 0; j < 10; ++j) acc[j] = 0.0f;
    for (int k = tid; k < 768; k += 256) {
        float xv = x[n * 768 + k];
#pragma unroll
        for (int j = 0; j < 10; ++j) acc[j] += xv * W1[k * 10 + j];
    }
#pragma unroll
    for (int j = 0; j < 10; ++j)
        for (int s = 1; s < 32; s <<= 1) acc[j] += __shfl_xor(acc[j], s, 32);
    __shared__ float red[8][10];
    int wave = tid >> 5, lane = tid & 31;
    if (lane == 0)
        for (int j = 0; j < 10; ++j) red[wave][j] = acc[j];
    __syncthreads();
    if (tid < 10) {
        float s = 0.0f;
        for (int w = 0; w < 8; ++w) s += red[w][tid];
        h1[n * 10 + tid] = s;
    }
}

// ---------------------------------------------------------------------------
// Kernel 5: layer-1 normalized scatter-add (edges + self loops), 10 feats
// ---------------------------------------------------------------------------
__global__ void k_agg1(const int* __restrict__ eidx, int E, int N,
                       const float* __restrict__ dinv, const float* __restrict__ h1,
                       float* __restrict__ agg1) {
    int e = blockIdx.x * blockDim.x + threadIdx.x;
    int total = E + N;
    if (e >= total) return;
    int s, d;
    if (e < E) { s = eidx[e]; d = eidx[E + e]; } else { s = d = e - E; }
    float w = dinv[s] * dinv[d];
    const float* hs = h1 + s * 10;
    float* ad = agg1 + d * 10;
#pragma unroll
    for (int j = 0; j < 10; ++j) atomicAdd(&ad[j], hs[j] * w);
}

// ---------------------------------------------------------------------------
// Kernel 6: h2 = relu(agg1 + b1) @ W2   (15000x10 @ 10x128)
// ---------------------------------------------------------------------------
__global__ void k_h2(const float* __restrict__ agg1, const float* __restrict__ b1,
                     const float* __restrict__ W2, float* __restrict__ h2) {
    int n = blockIdx.x;
    int j = threadIdx.x;                // 128 threads
    float acc = 0.0f;
#pragma unroll
    for (int k = 0; k < 10; ++k) {
        float hv = agg1[n * 10 + k] + b1[k];
        hv = hv > 0.0f ? hv : 0.0f;
        acc += hv * W2[k * 128 + j];
    }
    h2[n * 128 + j] = acc;
}

// ---------------------------------------------------------------------------
// Kernel 7: layer-2 normalized scatter-add, 128 feats, thread per (edge,j)
// ---------------------------------------------------------------------------
__global__ void k_agg2(const int* __restrict__ eidx, int E, int N,
                       const float* __restrict__ dinv, const float* __restrict__ h2,
                       float* __restrict__ agg2) {
    long long idx = (long long)blockIdx.x * blockDim.x + threadIdx.x;
    long long total = (long long)(E + N) * 128;
    if (idx >= total) return;
    int e = (int)(idx >> 7);
    int j = (int)(idx & 127);
    int s, d;
    if (e < E) { s = eidx[e]; d = eidx[E + e]; } else { s = d = e - E; }
    float w = dinv[s] * dinv[d];
    atomicAdd(&agg2[d * 128 + j], h2[s * 128 + j] * w);
}

// ---------------------------------------------------------------------------
// Kernel 8: fused gather + bf16 WMMA GEMM (i,g,o gates only; f unused) +
//           LSTM nonlinearity + reduction -> score.
// Block = 256 threads (8 waves), handles 16 triples. K = 1024, N_used = 384.
// ---------------------------------------------------------------------------
#define FSTR 1032   // 1024 + 8 pad (bf16 elems) -> conflict-free ds_load_b128

__global__ void k_lstm_score(const float* __restrict__ agg2, const float* __restrict__ b2,
                             const float* __restrict__ r_emb,
                             const unsigned short* __restrict__ Wb,
                             const float* __restrict__ b_ih, const float* __restrict__ b_hh,
                             const int* __restrict__ triples, int T,
                             float* __restrict__ score) {
    __shared__ unsigned short feat[16 * FSTR];
    __shared__ int headIdx[16], tailIdx[16];
    __shared__ float scoreAcc[16];

    int tid = threadIdx.x;
    int tileBase = blockIdx.x * 16;

    if (tid < 16) {
        int t = tileBase + tid;
        int tt = (t < T) ? t : 0;
        headIdx[tid] = triples[tt * 3 + 0];
        tailIdx[tid] = triples[tt * 3 + 2];
        scoreAcc[tid] = 0.0f;
    }
    __syncthreads();

    // Stage feat tile: [ z[head](128) | r_emb(768) | z[tail](128) ] as bf16.
    // z = agg2 + b2 fused here.
    for (int idx = tid; idx < 16 * 1024; idx += 256) {
        int m = idx >> 10;
        int k = idx & 1023;
        int t = tileBase + m;
        if (t >= T) t = 0;
        float v;
        if (k < 128)       v = agg2[(long long)headIdx[m] * 128 + k] + b2[k];
        else if (k < 896)  v = r_emb[(long long)t * 768 + (k - 128)];
        else               v = agg2[(long long)tailIdx[m] * 128 + (k - 896)] + b2[k - 896];
        feat[m * FSTR + k] = f2bf(v);
    }
    __syncthreads();

    // WMMA: wave w owns 16-column tile w of each of gates i (0..127),
    // g (256..383), o (384..511). A fragment shared across the 3 gates.
    int lane = tid & 31;
    int wave = tid >> 5;
    int half = lane >> 4;      // K-half selector (this code's contiguous k = 16*half + s, for A and B)
    int nl   = lane & 15;      // row (A) / column (B,C) within tile

    int ni = wave * 16 + nl;         // i-gate column
    int ng = 256 + wave * 16 + nl;   // g-gate column
    int no = 384 + wave * 16 + nl;   // o-gate column
    const unsigned short* rowI = Wb + (long long)ni * 1024;
    const unsigned short* rowG = Wb + (long long)ng * 1024;
    const unsigned short* rowO = Wb + (long long)no * 1024;
    const unsigned short* arow = &feat[nl * FSTR + half * 16];

    v8f ci = {}, cg = {}, co = {};
    for (int kb = 0; kb < 1024; kb += 32) {
        ABfrag a;
        a.h[0] = *(const v8bf*)(arow + kb);        // K = kb + half*16 .. +7
        a.h[1] = *(const v8bf*)(arow + kb + 8);    // K = kb + half*16 + 8 .. +15
        int kB = kb + half * 16;                   // B: 16 contiguous K per lane
        v16bf bI = *(const v16bf*)(rowI + kB);
        v16bf bG = *(const v16bf*)(rowG + kB);
        v16bf bO = *(const v16bf*)(rowO + kB);
        ci = __builtin_amdgcn_wmma_f32_16x16x32_bf16(false, a.v, false, bI, (short)0, ci, false, false);
        cg = __builtin_amdgcn_wmma_f32_16x16x32_bf16(false, a.v, false, bG, (short)0, cg, false, false);
        co = __builtin_amdgcn_wmma_f32_16x16x32_bf16(false, a.v, false, bO, (short)0, co, false, false);
    }

    float biasI = b_ih[ni] + b_hh[ni];
    float biasG = b_ih[ng] + b_hh[ng];
    float biasO = b_ih[no] + b_hh[no];

    // C layout: VGPR r holds D[r + 8*half][nl]. LSTM elementwise + reduce over n.
#pragma unroll
    for (int r = 0; r < 8; ++r) {
        float iv = ci[r] + biasI;
        float gv = cg[r] + biasG;
        float ov = co[r] + biasO;
        float cc   = sigm(iv) * tanhf(gv);
        float outv = sigm(ov) * tanhf(cc);
        // sum over the 16 lanes sharing this m (masks <16 stay in half)
        for (int s = 1; s < 16; s <<= 1) outv += __shfl_xor(outv, s, 32);
        if (nl == 0) atomicAdd(&scoreAcc[r + 8 * half], outv);
    }
    __syncthreads();

    if (tid < 16) {
        int t = tileBase + tid;
        if (t < T) score[t] = sigm(scoreAcc[tid]);
    }
}

// ---------------------------------------------------------------------------
// Host launcher
// ---------------------------------------------------------------------------
extern "C" void kernel_launch(void* const* d_in, const int* in_sizes, int n_in,
                              void* d_out, int out_size, void* d_ws, size_t ws_size,
                              hipStream_t stream) {
    const float* x     = (const float*)d_in[0];
    const float* r_emb = (const float*)d_in[1];
    const float* W1    = (const float*)d_in[2];
    const float* b1    = (const float*)d_in[3];
    const float* W2    = (const float*)d_in[4];
    const float* b2    = (const float*)d_in[5];
    const float* W_ih  = (const float*)d_in[6];
    // d_in[7] = W_hh: unused by the reference math
    const float* b_ih  = (const float*)d_in[8];
    const float* b_hh  = (const float*)d_in[9];
    const int* eidx    = (const int*)d_in[10];
    const int* triples = (const int*)d_in[11];

    int N = in_sizes[0] / 768;     // 15000
    int E = in_sizes[10] / 2;      // 200000
    int T = in_sizes[11] / 3;      // 100000

    // Workspace carve-out (256B aligned)
    char* ws = (char*)d_ws;
    size_t off = 0;
    auto carve = [&](size_t bytes) -> void* {
        off = (off + 255) & ~(size_t)255;
        void* p = ws + off;
        off += bytes;
        return p;
    };
    float* deg            = (float*)carve((size_t)N * 4);          // later holds dinv
    float* h1             = (float*)carve((size_t)N * 10 * 4);
    float* agg1           = (float*)carve((size_t)N * 10 * 4);
    float* h2             = (float*)carve((size_t)N * 128 * 4);
    float* agg2           = (float*)carve((size_t)N * 128 * 4);
    unsigned short* Wb    = (unsigned short*)carve((size_t)512 * 1024 * 2);
    float* score = (float*)d_out;

    // 0) init
    {
        int tot = N * 128;
        k_init<<<(tot + 255) / 256, 256, 0, stream>>>(deg, agg1, agg2, N);
    }
    // 1) W_ih -> bf16
    {
        int tot = 512 * 1024;
        k_w2bf<<<(tot + 255) / 256, 256, 0, stream>>>(W_ih, Wb, tot);
    }
    // 2) degree
    k_deg<<<(E + 255) / 256, 256, 0, stream>>>(eidx, E, deg);
    // 3) dinv
    k_dinv<<<(N + 255) / 256, 256, 0, stream>>>(deg, N);
    // 4) h1 = x @ W1
    k_xw1<<<N, 256, 0, stream>>>(x, W1, h1);
    // 5) layer-1 aggregate
    k_agg1<<<(E + N + 255) / 256, 256, 0, stream>>>(eidx, E, N, deg, h1, agg1);
    // 6) h2 = relu(agg1 + b1) @ W2
    k_h2<<<N, 128, 0, stream>>>(agg1, b1, W2, h2);
    // 7) layer-2 aggregate
    {
        long long tot = (long long)(E + N) * 128;
        int blocks = (int)((tot + 255) / 256);
        k_agg2<<<blocks, 256, 0, stream>>>(eidx, E, N, deg, h2, agg2);
    }
    // 8) fused gather + WMMA GEMM + LSTM + score
    k_lstm_score<<<(T + 15) / 16, 256, 0, stream>>>(agg2, b2, r_emb, Wb,
                                                    b_ih, b_hh, triples, T, score);
}
